// AdaptiveSinLayer_11982958756456
// MI455X (gfx1250) — matrix operand order repaired and checked by reading, not compiled, not run on hardware
//
#include <hip/hip_runtime.h>
#include <hip/hip_fp16.h>

typedef __attribute__((ext_vector_type(16))) _Float16 v16h;
typedef __attribute__((ext_vector_type(8)))  _Float16 v8h;
typedef __attribute__((ext_vector_type(8)))  float    v8f;
typedef __attribute__((ext_vector_type(4)))  float    v4f;

#define IN_F    256
#define OUT_F   256
#define PTS     256
#define N_TILES 1024
#define OMEGA   30.0f

// LDS strides (in halves). lds_w padded to 40 so each column base is 16B-aligned
// (40 = 5*8 halves) and store bank conflicts are limited to 2-way.
#define LDSA_STRIDE 32
#define LDSW_STRIDE 40

// One block = 256 threads = 8 waves (wave32). Block computes a 64-row x 256-col
// slab of one tile's output. Grid = N_TILES * (PTS/64) = 4096.
// 2-stage software pipeline: global fetch of chunk k+1 overlaps WMMA of chunk k;
// LDS is double-buffered (48 KB of the 320 KB WGP pool).
__global__ __launch_bounds__(256) void siren_wmma_f16(
    const float* __restrict__ x,     // [T, 256, 256]
    const float* __restrict__ w,     // [C, 256, 256]
    const float* __restrict__ bias,  // [256]
    const int*   __restrict__ idx,   // [T]
    float* __restrict__ out)         // [T, 256, 256]
{
    __shared__ __align__(16) _Float16 lds_a[2][64 * LDSA_STRIDE];   //  2 x 4 KB
    __shared__ __align__(16) _Float16 lds_w[2][256 * LDSW_STRIDE];  //  2 x 20 KB

    const int tid  = threadIdx.x;
    const int lane = tid & 31;
    const int wv   = tid >> 5;          // wave 0..7
    const int tile = blockIdx.x >> 2;
    const int mblk = blockIdx.x & 3;
    const int m0   = mblk * 64;

    const int ch = idx[tile];

    const float* Ag = x + (size_t)tile * (PTS * IN_F) + (size_t)m0 * IN_F;
    const float* Wg = w + (size_t)ch * (IN_F * OUT_F);

    const int mt = wv & 3;              // m-tile within the 64-row slab (16 rows)
    const int ng = wv >> 2;             // n-group: 0 -> cols 0..127, 1 -> 128..255
    const int r  = lane & 15;
    const int c  = lane >> 4;

    // staging geometry (per thread)
    const int arow = tid >> 2;
    const int acol = (tid & 3) * 8;

    // ---- pipeline register state ----
    v4f   pf_a0, pf_a1;                 // 8 floats of the A chunk
    float pf_w[32];                     // one W column slice (k=0..31 at n=tid)

    auto fetch = [&](int kc) {
        const float* asrc = Ag + (size_t)arow * IN_F + kc + acol;
        pf_a0 = *(const v4f*)(asrc);
        pf_a1 = *(const v4f*)(asrc + 4);
#pragma unroll
        for (int e = 0; e < 32; ++e)
            pf_w[e] = Wg[(size_t)(kc + e) * OUT_F + tid];
    };

    auto stage = [&](int buf) {
        _Float16* da = &lds_a[buf][arow * LDSA_STRIDE + acol];
#pragma unroll
        for (int e = 0; e < 4; ++e) da[e]     = (_Float16)pf_a0[e];
#pragma unroll
        for (int e = 0; e < 4; ++e) da[4 + e] = (_Float16)pf_a1[e];
        _Float16* dw = &lds_w[buf][tid * LDSW_STRIDE];
#pragma unroll
        for (int e = 0; e < 32; ++e) dw[e] = (_Float16)pf_w[e];
    };

    auto loadB = [&](int buf, int j) -> v16h {
        // B fragment (32x16): lane half c holds K = c*8+0..7 then K = 16+c*8+0..7, N=lane%16;
        // transposed LDS makes this two aligned b128 loads of 8 contiguous halves.
        const int n = ng * 128 + j * 16 + r;
        const _Float16* bb = &lds_w[buf][n * LDSW_STRIDE + c * 8];
        v8h blo = *(const v8h*)(bb);
        v8h bhi = *(const v8h*)(bb + 16);
        return __builtin_shufflevector(blo, bhi,
                 0,1,2,3,4,5,6,7,8,9,10,11,12,13,14,15);
    };

    v8f acc[8] = {};                    // 8 n-tiles of 16x16 f32 accumulators

    // ---- prologue: stage chunk 0 ----
    fetch(0);
    stage(0);
    __syncthreads();

    int buf = 0;
#pragma unroll 1
    for (int i = 0; i < IN_F / 32; ++i) {
        const bool more = (i < IN_F / 32 - 1);
        if (more) fetch((i + 1) * 32);   // overlap global fetch with compute

        // ---- A fragment (ISA 16-bit A 16x32 layout) ----
        // lane half c: VGPR0-3 hold K = c*8+0..7, VGPR4-7 hold K = 16+c*8+0..7
        const _Float16* ab = &lds_a[buf][(mt * 16 + r) * LDSA_STRIDE + c * 8];
        v8h alo = *(const v8h*)(ab);
        v8h ahi = *(const v8h*)(ab + 16);
        v16h a = __builtin_shufflevector(alo, ahi,
                 0,1,2,3,4,5,6,7,8,9,10,11,12,13,14,15);

        // B fragments pipelined by one so ds_load overlaps the WMMA.
        v16h bcur = loadB(buf, 0);
#pragma unroll
        for (int j = 0; j < 8; ++j) {
            v16h bnext;
            if (j < 7) bnext = loadB(buf, j + 1);
            acc[j] = __builtin_amdgcn_wmma_f32_16x16x32_f16(
                         /*neg_a=*/false, a, /*neg_b=*/false, bcur,
                         /*c_mod=*/(short)0, acc[j],
                         /*reuse_a=*/false, /*reuse_b=*/false);
            bcur = bnext;
        }

        if (more) {
            stage(buf ^ 1);              // convert + store next chunk
            __syncthreads();
            buf ^= 1;
        }
    }

    // ---- epilogue: C layout -> M = i + 8*c, N = n0 + r per VGPR i ----
    const int mrow = m0 + mt * 16 + 8 * c;
#pragma unroll
    for (int j = 0; j < 8; ++j) {
        const int n = ng * 128 + j * 16 + r;
        const float bv = bias[n];
        float* dst = out + (size_t)tile * (PTS * OUT_F) + (size_t)mrow * OUT_F + n;
#pragma unroll
        for (int i = 0; i < 8; ++i) {
            dst[(size_t)i * OUT_F] = __sinf(OMEGA * (acc[j][i] + bv));
        }
    }
}

extern "C" void kernel_launch(void* const* d_in, const int* in_sizes, int n_in,
                              void* d_out, int out_size, void* d_ws, size_t ws_size,
                              hipStream_t stream) {
    const float* x    = (const float*)d_in[0];   // [1024,256,256] f32
    const float* wgt  = (const float*)d_in[1];   // [1024,256,256] f32
    const float* bias = (const float*)d_in[2];   // [256] f32
    const int*   idx  = (const int*)d_in[3];     // [1024] int
    float* out = (float*)d_out;

    dim3 grid(N_TILES * (PTS / 64));             // 4096 blocks
    dim3 block(256);                             // 8 waves (wave32)
    siren_wmma_f16<<<grid, block, 0, stream>>>(x, wgt, bias, idx, out);
}
